// QuantizedLinear_17042430231010
// MI455X (gfx1250) — 8-bit matrix operand order repaired and checked by reading, compiled, not run on hardware
//
#include <hip/hip_runtime.h>

typedef __attribute__((ext_vector_type(8))) int v8i;

#if __has_builtin(__builtin_amdgcn_global_load_async_to_lds_b128)
#define HAS_ASYNC_LDS 1
#else
#define HAS_ASYNC_LDS 0
#endif

typedef int vi4 __attribute__((vector_size(16)));
typedef __attribute__((address_space(1))) vi4 global_vi4;
typedef __attribute__((address_space(3))) vi4 local_vi4;

__device__ __forceinline__ void copy16_g2l(const signed char* g, signed char* l) {
#if HAS_ASYNC_LDS
    __builtin_amdgcn_global_load_async_to_lds_b128(
        (global_vi4*)g, (local_vi4*)l,  0,  0);
#else
    *(int4*)l = *(const int4*)g;
#endif
}

template <int N>
__device__ __forceinline__ void wait_async() {
#if HAS_ASYNC_LDS
#if __has_builtin(__builtin_amdgcn_s_wait_asynccnt)
    __builtin_amdgcn_s_wait_asynccnt(N);
#else
    asm volatile("s_wait_asynccnt %0" ::"i"(N) : "memory");
#endif
#endif
}

__device__ __forceinline__ int clamp_q(int q) {
    return q < -128 ? -128 : (q > 127 ? 127 : q);
}

__device__ __forceinline__ int pack4(float a, float b, float c, float d, float inv) {
    int q0 = clamp_q(__float2int_rn(a * inv));
    int q1 = clamp_q(__float2int_rn(b * inv));
    int q2 = clamp_q(__float2int_rn(c * inv));
    int q3 = clamp_q(__float2int_rn(d * inv));
    return (q0 & 255) | ((q1 & 255) << 8) | ((q2 & 255) << 16) | ((q3 & 255) << 24);
}

__global__ __launch_bounds__(256)
void quant_x_kernel(const float* __restrict__ x, int* __restrict__ qx,
                    const float* __restrict__ act_scale, int n4) {
    int i = blockIdx.x * 256 + threadIdx.x;
    if (i >= n4) return;
    float inv = 1.0f / act_scale[0];
    float4 v = reinterpret_cast<const float4*>(x)[i];
    qx[i] = pack4(v.x, v.y, v.z, v.w, inv);
}

__global__ __launch_bounds__(256)
void quant_w_kernel(const float* __restrict__ w, int* __restrict__ qw,
                    const float* __restrict__ wscale, int K4) {
    int row = blockIdx.y;
    int i   = blockIdx.x * 256 + threadIdx.x;
    float inv = 1.0f / wscale[row];
    size_t idx = (size_t)row * K4 + i;
    float4 v = reinterpret_cast<const float4*>(w)[idx];
    qw[idx] = pack4(v.x, v.y, v.z, v.w, inv);
}

#define BLOCK_M 64
#define BLOCK_N 256
#define TILE_K  64
#define A_STRIDE 80

__global__ __launch_bounds__(256)
void gemm_iu8_kernel(const signed char* __restrict__ qx,
                     const signed char* __restrict__ qw,
                     const float* __restrict__ wscale,
                     const float* __restrict__ bias,
                     const float* __restrict__ act_scale,
                     float* __restrict__ out,
                     int M, int N, int K) {
    __shared__ __align__(16) signed char ldsA[2][BLOCK_M * A_STRIDE];

    const int tid  = threadIdx.x;
    const int lane = tid & 31;
    const int wid  = tid >> 5;
    const int l16  = lane & 15;
    const int h    = lane >> 4;
    const int blockM = blockIdx.y * BLOCK_M;
    const int mLoc   = (wid >> 2) * 32;
    const int nBase  = blockIdx.x * BLOCK_N + (wid & 3) * 64;

    const int cRow = tid >> 2;
    const int cOff = (tid & 3) * 16;
    const signed char* gA = qx + (size_t)(blockM + cRow) * K + cOff;
    signed char* lA0 = &ldsA[0][cRow * A_STRIDE + cOff];
    signed char* lA1 = &ldsA[1][cRow * A_STRIDE + cOff];

    v8i acc[2][4];
#pragma unroll
    for (int mt = 0; mt < 2; ++mt)
#pragma unroll
        for (int nt = 0; nt < 4; ++nt)
#pragma unroll
            for (int e = 0; e < 8; ++e) acc[mt][nt][e] = 0;

    const signed char* bPtr[4];
#pragma unroll
    for (int nt = 0; nt < 4; ++nt)
        bPtr[nt] = qw + (size_t)(nBase + nt * 16 + l16) * K + h * 16;

    const int nK = K / TILE_K;

    copy16_g2l(gA, lA0);

    for (int i = 0; i < nK; ++i) {
        if (i + 1 < nK) {
            copy16_g2l(gA + (size_t)(i + 1) * TILE_K, ((i + 1) & 1) ? lA1 : lA0);
            wait_async<1>();
        } else {
            wait_async<0>();
        }
        __syncthreads();

        const signed char* base = &ldsA[i & 1][0];
        v8i a[2];
#pragma unroll
        for (int mt = 0; mt < 2; ++mt) {
            const signed char* p =
                base + (mLoc + mt * 16 + l16) * A_STRIDE + h * 16;
            int4 t0 = *(const int4*)(p);
            int4 t1 = *(const int4*)(p + 32);
            a[mt][0] = t0.x; a[mt][1] = t0.y; a[mt][2] = t0.z; a[mt][3] = t0.w;
            a[mt][4] = t1.x; a[mt][5] = t1.y; a[mt][6] = t1.z; a[mt][7] = t1.w;
        }

        const int k0 = i * TILE_K;
        v8i b[4];
#pragma unroll
        for (int nt = 0; nt < 4; ++nt) {
            const signed char* p = bPtr[nt] + k0;
            int4 u0 = *(const int4*)(p);
            int4 u1 = *(const int4*)(p + 32);
            b[nt][0] = u0.x; b[nt][1] = u0.y; b[nt][2] = u0.z; b[nt][3] = u0.w;
            b[nt][4] = u1.x; b[nt][5] = u1.y; b[nt][6] = u1.z; b[nt][7] = u1.w;
        }

#pragma unroll
        for (int mt = 0; mt < 2; ++mt)
#pragma unroll
            for (int nt = 0; nt < 4; ++nt)
                acc[mt][nt] = __builtin_amdgcn_wmma_i32_16x16x64_iu8(
                     true, a[mt],  true, b[nt],
                    acc[mt][nt],  false,  false);

        __syncthreads();
    }

    const float as = act_scale[0];
#pragma unroll
    for (int nt = 0; nt < 4; ++nt) {
        const int col = nBase + nt * 16 + l16;
        const float sc = as * wscale[col];
        const float bv = bias[col];
#pragma unroll
        for (int mt = 0; mt < 2; ++mt) {
            const int rowBase = blockM + mLoc + mt * 16 + 8 * h;
#pragma unroll
            for (int r = 0; r < 8; ++r)
                out[(size_t)(rowBase + r) * N + col] =
                    (float)acc[mt][nt][r] * sc + bv;
        }
    }
}

extern "C" void kernel_launch(void* const* d_in, const int* in_sizes, int n_in,
                              void* d_out, int out_size, void* d_ws, size_t ws_size,
                              hipStream_t stream) {
    const float* x         = (const float*)d_in[0];
    const float* w         = (const float*)d_in[1];
    const float* bias      = (const float*)d_in[2];
    const float* act_scale = (const float*)d_in[3];
    const float* wscale    = (const float*)d_in[4];

    const int nX = in_sizes[0];
    const int nW = in_sizes[1];
    const int N  = in_sizes[2];
    const int K  = nW / N;
    const int M  = nX / K;

    signed char* qx = (signed char*)d_ws;
    signed char* qw = (signed char*)d_ws + (size_t)nX;

    quant_x_kernel<<<(nX / 4 + 255) / 256, 256, 0, stream>>>(
        x, (int*)qx, act_scale, nX / 4);

    dim3 gw(K / 4 / 256, N);
    quant_w_kernel<<<gw, 256, 0, stream>>>(w, (int*)qw, wscale, K / 4);

    dim3 gg(N / BLOCK_N, M / BLOCK_M);
    gemm_iu8_kernel<<<gg, 256, 0, stream>>>(qx, qw, wscale, bias, act_scale,
                                            (float*)d_out, M, N, K);
}
